// MACUnit_9612136809197
// MI455X (gfx1250) — matrix operand order repaired and checked by reading, not compiled, not run on hardware
//
#include <hip/hip_runtime.h>
#include <hip/hip_bf16.h>

typedef __bf16 bf16;
typedef __attribute__((ext_vector_type(16))) __bf16 v16bf;
typedef __attribute__((ext_vector_type(8)))  __bf16 v8bf;
typedef __attribute__((ext_vector_type(8)))  float   v8f;

#define DIMD  512
#define BATCH 512
#define LCTX  32
#define HWN   196
#define KSLOT 8
#define NSTEP 4
#define CHROWS 32
#define NCHUNK (DIMD/CHROWS)

#if defined(__has_builtin)
# if __has_builtin(__builtin_amdgcn_tensor_load_to_lds)
#  define HAVE_TDM 1
# endif
#endif
#ifndef HAVE_TDM
# define HAVE_TDM 0
#endif

__device__ __forceinline__ float elu1(float x){ return x > 0.f ? x : (__expf(x) - 1.f); }

__device__ __forceinline__ bf16 to_bf16(float f){
  unsigned u = __float_as_uint(f);
  unsigned r = ((u >> 16) & 1u) + 0x7fffu;            // round-to-nearest-even
  unsigned short s = (unsigned short)((u + r) >> 16);
  return __builtin_bit_cast(bf16, s);
}

#if HAVE_TDM
// Issue a 2D TDM load: `rows` x HWN f32 tile from global into LDS (D# per ISA §8).
// This toolchain exposes the 6-arg builtin form:
//   (uint32x4 g0, int32x8 g1, int32x4 g2, int32x4 g3, int32x8 g4, i32 cpol)
__device__ __forceinline__ void tdm_load_2d(unsigned lds_addr, const void* gaddr, int rows){
  typedef __attribute__((ext_vector_type(4))) unsigned v4u_;
  typedef __attribute__((ext_vector_type(8))) int      v8i_;
  typedef __attribute__((ext_vector_type(4))) int      v4i_;
  unsigned long long ga = (unsigned long long)(uintptr_t)gaddr;
  v4u_ g0;
  g0[0] = 1u;                                             // count=1 (valid user D#)
  g0[1] = lds_addr;                                       // lds_addr [63:32]
  g0[2] = (unsigned)(ga & 0xffffffffu);                   // global_addr[31:0]
  g0[3] = (unsigned)((ga >> 32) & 0x01ffffffu) | (2u<<30);// global_addr[56:32] | type=2
  v8i_ g1;
  g1[0] = (2 << 16);                                      // data_size=4B, no multicast
  g1[1] = (int)((HWN & 0xffff) << 16);                    // tensor_dim0[15:0]
  g1[2] = (int)((HWN >> 16) | ((DIMD & 0xffff) << 16));   // tensor_dim0[31:16]|tensor_dim1[15:0]
  g1[3] = (int)((DIMD >> 16) | (HWN << 16));              // tensor_dim1[31:16]|tile_dim0
  g1[4] = rows & 0xffff;                                  // tile_dim1 | tile_dim2=0
  g1[5] = HWN;                                            // tensor_dim0_stride[31:0]
  g1[6] = 0;
  g1[7] = 0;
  v4i_ z4 = {0,0,0,0};
  v8i_ z8 = {0,0,0,0,0,0,0,0};
  __builtin_amdgcn_tensor_load_to_lds(g0, g1, z4, z4, z8, 0);
}
#endif

// ---------------- conversion kernels ----------------
__global__ void k_cvt_bf16(bf16* __restrict__ dst, const float* __restrict__ src, int n){
  int i = blockIdx.x*blockDim.x + threadIdx.x;
  if (i < n) dst[i] = to_bf16(src[i]);
}

__global__ void k_cvt_bf16_strided(bf16* __restrict__ dst, int ldd,
                                   const float* __restrict__ src, int lds_,
                                   int rows, int cols){
  int i = blockIdx.x*blockDim.x + threadIdx.x;
  if (i < rows*cols){
    int r = i / cols, c = i % cols;
    dst[(size_t)r*ldd + c] = to_bf16(src[(size_t)r*lds_ + c]);
  }
}

// src: R x C f32 row-major -> dst: C x R bf16 row-major (dst[c,r] = src[r,c])
__global__ void k_cvt_bf16_T(bf16* __restrict__ dst, const float* __restrict__ src, int R, int C){
  int i = blockIdx.x*blockDim.x + threadIdx.x;
  if (i < R*C){ int r = i / C, c = i % C; dst[(size_t)c*R + r] = to_bf16(src[i]); }
}

__global__ void k_make_v(bf16* __restrict__ v, const float* __restrict__ control,
                         const float* __restrict__ aw, int n){
  int i = blockIdx.x*blockDim.x + threadIdx.x;
  if (i < n) v[i] = to_bf16(control[i] * aw[i & (DIMD-1)]);
}

// ---------------- WMMA bf16 GEMM: C = A(MxK) @ W(NxK)^T + bias ----------------
// One wave per 16x64 strip (4 N-tiles reuse one A fragment). K-loop of 32.
// Operands are L2-resident (<=2MB) -> direct global b128 loads feed v_wmma.
__global__ __launch_bounds__(256) void k_gemm_bf16(
  const bf16* __restrict__ A, int lda,
  const bf16* __restrict__ W,
  const float* __restrict__ bias,
  float* __restrict__ Cf, bf16* __restrict__ Cb, int ldc,
  int M, int N, int K, int act)
{
  int stripsN = N >> 6;                       // N in 64-wide strips
  int strip = blockIdx.x*8 + (threadIdx.x >> 5);
  if (strip >= (M >> 4)*stripsN) return;
  int lane = threadIdx.x & 31;
  int tm  = (strip / stripsN) << 4;
  int tn0 = (strip % stripsN) << 6;
  int half = lane >> 4, l15 = lane & 15;
  // A layout (16-bit, 16x32): lane halves split K: half0 -> K 0..15 ; half1 -> K 16..31
  const bf16* Arow = A + (size_t)(tm + l15)*lda + 16*half;
  // B layout (32x16): lane = N, half selects K 0..15 / 16..31, pairs packed per VGPR (contiguous)
  const bf16* W0 = W + (size_t)(tn0      + l15)*K + 16*half;
  const bf16* W1 = W + (size_t)(tn0 + 16 + l15)*K + 16*half;
  const bf16* W2 = W + (size_t)(tn0 + 32 + l15)*K + 16*half;
  const bf16* W3 = W + (size_t)(tn0 + 48 + l15)*K + 16*half;
  v8f acc0 = {}, acc1 = {}, acc2 = {}, acc3 = {};
  for (int k0 = 0; k0 < K; k0 += 32){
    union { v16bf v; v8bf h[2]; } a;
    a.h[0] = *(const v8bf*)(Arow + k0);
    a.h[1] = *(const v8bf*)(Arow + k0 + 8);
    v16bf b0 = *(const v16bf*)(W0 + k0);
    v16bf b1 = *(const v16bf*)(W1 + k0);
    v16bf b2 = *(const v16bf*)(W2 + k0);
    v16bf b3 = *(const v16bf*)(W3 + k0);
    acc0 = __builtin_amdgcn_wmma_f32_16x16x32_bf16(false, a.v, false, b0, (short)0, acc0, false, false);
    acc1 = __builtin_amdgcn_wmma_f32_16x16x32_bf16(false, a.v, false, b1, (short)0, acc1, false, false);
    acc2 = __builtin_amdgcn_wmma_f32_16x16x32_bf16(false, a.v, false, b2, (short)0, acc2, false, false);
    acc3 = __builtin_amdgcn_wmma_f32_16x16x32_bf16(false, a.v, false, b3, (short)0, acc3, false, false);
  }
  int rbase = tm + 8*half;   // C: lane half selects M 0..7 / 8..15, VGPR r = row
  v8f accs[4] = {acc0, acc1, acc2, acc3};
  #pragma unroll
  for (int t = 0; t < 4; ++t){
    int col = tn0 + t*16 + l15;
    float bv = bias ? bias[col] : 0.f;
    #pragma unroll
    for (int r = 0; r < 8; ++r){
      float x = accs[t][r] + bv;
      if (act == 1) x = elu1(x);
      size_t idx = (size_t)(rbase + r)*ldc + col;
      if (Cf) Cf[idx] = x;
      if (Cb) Cb[idx] = to_bf16(x);
    }
  }
}

// ---------------- context attention (one block per batch) ----------------
__global__ __launch_bounds__(256) void k_ctx_attn(
  const float* __restrict__ cqi, const float* __restrict__ context,
  const float* __restrict__ attn_w, float* __restrict__ control,
  bf16* __restrict__ cq0, bf16* __restrict__ lr0, bf16* __restrict__ lrh0)
{
  int b = blockIdx.x, tid = threadIdx.x;
  __shared__ float s_qa[DIMD];
  __shared__ float s_red[256];
  __shared__ float s_p[LCTX];
  const float* q = cqi + (size_t)b*DIMD;
  for (int dd = tid; dd < DIMD; dd += 256) s_qa[dd] = q[dd]*attn_w[dd];
  __syncthreads();
  const float* ctx = context + (size_t)b*LCTX*DIMD;
  int l = tid >> 3, sub = tid & 7;
  const float* row = ctx + (size_t)l*DIMD;
  float part = 0.f;
  for (int dd = sub; dd < DIMD; dd += 8) part = fmaf(s_qa[dd], row[dd], part);
  s_red[tid] = part; __syncthreads();
  if (sub == 0){
    float sc = 0.f;
    for (int j = 0; j < 8; ++j) sc += s_red[tid + j];
    s_p[l] = sc;
  }
  __syncthreads();
  if (tid == 0){
    float m = -1e30f;
    for (int j = 0; j < LCTX; ++j) m = fmaxf(m, s_p[j]);
    float s = 0.f;
    for (int j = 0; j < LCTX; ++j){ float e = __expf(s_p[j]-m); s_p[j] = e; s += e; }
    float inv = 1.f/s;
    for (int j = 0; j < LCTX; ++j) s_p[j] *= inv;
  }
  __syncthreads();
  for (int dd = tid; dd < DIMD; dd += 256){
    float c = 0.f;
    for (int j = 0; j < LCTX; ++j) c = fmaf(s_p[j], ctx[(size_t)j*DIMD + dd], c);
    control[(size_t)b*DIMD + dd] = c;
    bf16 cb = to_bf16(c);
    cq0 [(size_t)b*(2*DIMD) + dd] = cb;
    lr0 [(size_t)b*(2*DIMD) + dd] = cb;
    lrh0[(size_t)b*(2*DIMD) + dd] = cb;
  }
}

// ---------------- KB attention: the HBM-bound streamer (one block per batch) ----
// Score pass stages kb_proj through LDS with double-buffered TDM loads
// (tensor_load_to_lds + s_wait_tensorcnt) to overlap DMA with FMA consumption.
__global__ __launch_bounds__(256) void k_kb_attn(
  const float* __restrict__ u, const float* __restrict__ mem_p,
  const float* __restrict__ kb, const float* __restrict__ kn,
  float* __restrict__ readv, bf16* __restrict__ lr_rd)
{
  int b = blockIdx.x, tid = threadIdx.x;
  __shared__ float s_kb[2][CHROWS*HWN];
  __shared__ float s_w[DIMD];
  __shared__ float s_red[256];
  __shared__ float s_p[256];
  for (int dd = tid; dd < DIMD; dd += 256)
    s_w[dd] = u[(size_t)b*2*DIMD + dd]*mem_p[(size_t)b*DIMD + dd]
            + u[(size_t)b*2*DIMD + DIMD + dd];
  const float* kbb = kb + (size_t)b*DIMD*HWN;
#if HAVE_TDM
  unsigned lds0 = (unsigned)(uintptr_t)(&s_kb[0][0]);
  unsigned lds1 = (unsigned)(uintptr_t)(&s_kb[1][0]);
  bool issuer = (tid >> 5) == 0;                 // wave 0 drives the TDM pipeline
  if (issuer) tdm_load_2d(lds0, kbb, CHROWS);
#endif
  float sc_acc = 0.f;
  for (int c = 0; c < NCHUNK; ++c){
#if HAVE_TDM
    if (issuer){
      if (c + 1 < NCHUNK){
        tdm_load_2d(((c+1)&1) ? lds1 : lds0,
                    kbb + (size_t)(c+1)*CHROWS*HWN, CHROWS);
        __builtin_amdgcn_s_wait_tensorcnt(1);    // oldest (chunk c) complete
      } else {
        __builtin_amdgcn_s_wait_tensorcnt(0);
      }
    }
    __syncthreads();                             // publish chunk c to all waves
#else
    {
      const float* g = kbb + (size_t)c*CHROWS*HWN;
      float* dst = &s_kb[c & 1][0];
      for (int i = tid; i < CHROWS*HWN; i += 256) dst[i] = g[i];
      __syncthreads();
    }
#endif
    if (tid < HWN){
      const float* base = &s_kb[c & 1][0] + tid;  // lanes -> consecutive DWORDs, no bank conflict
      const float* wv = &s_w[c*CHROWS];
      #pragma unroll
      for (int r = 0; r < CHROWS; ++r) sc_acc = fmaf(wv[r], base[(size_t)r*HWN], sc_acc);
    }
    __syncthreads();                             // chunk consumed; buffer may be overwritten
  }
  float sc = (tid < HWN) ? sc_acc : -1e30f;
  s_red[tid] = sc; __syncthreads();
  for (int o = 128; o > 0; o >>= 1){ if (tid < o) s_red[tid] = fmaxf(s_red[tid], s_red[tid+o]); __syncthreads(); }
  float m = s_red[0]; __syncthreads();
  float e = (tid < HWN) ? __expf(sc - m) : 0.f;
  s_red[tid] = e; __syncthreads();
  for (int o = 128; o > 0; o >>= 1){ if (tid < o) s_red[tid] += s_red[tid+o]; __syncthreads(); }
  float inv = 1.f/s_red[0];
  s_p[tid] = e*inv;
  __syncthreads();
  const float* knb = kn + (size_t)b*DIMD*HWN;
  int wave = tid >> 5, lane = tid & 31;
  for (int dd = wave; dd < DIMD; dd += 8){
    const float* row = knb + (size_t)dd*HWN;      // contiguous per row, coalesced per wave
    float acc = 0.f;
    for (int hw = lane; hw < HWN; hw += 32) acc = fmaf(s_p[hw], row[hw], acc);
    for (int o = 16; o > 0; o >>= 1) acc += __shfl_xor(acc, o, 32);
    if (lane == 0){
      readv[(size_t)b*DIMD + dd] = acc;
      lr_rd[(size_t)b*(2*DIMD) + dd] = to_bf16(acc);
    }
  }
}

// ---------------- history attention (K=8; one block per batch) ----------------
__global__ __launch_bounds__(256) void k_hist_attn(
  const float* __restrict__ u, const float* __restrict__ memh,
  const float* __restrict__ hist, float* __restrict__ rvi,
  float* __restrict__ readh, bf16* __restrict__ lrh_rd)
{
  int b = blockIdx.x, tid = threadIdx.x;
  __shared__ float s_w[DIMD];
  __shared__ float s_red[256];
  __shared__ float s_sc[KSLOT];
  __shared__ float s_p[KSLOT];
  for (int dd = tid; dd < DIMD; dd += 256)
    s_w[dd] = u[(size_t)b*2*DIMD + dd]*memh[(size_t)b*DIMD + dd]
            + u[(size_t)b*2*DIMD + DIMD + dd];
  __syncthreads();
  const float* hb = hist + (size_t)b*DIMD*KSLOT;
  int k = tid & 7, g = tid >> 3;
  float part = 0.f;
  for (int dd = g; dd < DIMD; dd += 32) part = fmaf(s_w[dd], hb[(size_t)dd*KSLOT + k], part);
  s_red[tid] = part; __syncthreads();
  if (tid < KSLOT){
    float sc = 0.f;
    for (int gg = 0; gg < 32; ++gg) sc += s_red[gg*8 + tid];
    s_sc[tid] = sc;
  }
  __syncthreads();
  if (tid == 0){
    float m = -1e30f;
    for (int j = 0; j < KSLOT; ++j) m = fmaxf(m, s_sc[j]);
    float s = 0.f, e[KSLOT];
    for (int j = 0; j < KSLOT; ++j){ e[j] = __expf(s_sc[j]-m); s += e[j]; }
    float inv = 1.f/s;
    for (int j = 0; j < KSLOT; ++j){ s_p[j] = e[j]*inv; rvi[(size_t)b*KSLOT + j] = e[j]*inv; }
  }
  __syncthreads();
  for (int dd = tid; dd < DIMD; dd += 256){
    const float* hrow = hb + (size_t)dd*KSLOT;
    float a = 0.f;
    for (int j = 0; j < KSLOT; ++j) a = fmaf(s_p[j], hrow[j], a);
    readh[(size_t)b*DIMD + dd] = a;
    lrh_rd[(size_t)b*(2*DIMD) + dd] = to_bf16(a);
  }
}

// ---------------- sigmoid gate: sigmoid(h[b,:]·w2 + b2) ----------------
__global__ __launch_bounds__(256) void k_gate(const float* __restrict__ h,
                                              const float* __restrict__ w2,
                                              const float* __restrict__ b2,
                                              float* __restrict__ out, int n){
  int b = blockIdx.x, tid = threadIdx.x;
  __shared__ float s_red[256];
  const float* row = h + (size_t)b*n;
  float p = 0.f;
  for (int j = tid; j < n; j += 256) p = fmaf(row[j], w2[j], p);
  s_red[tid] = p; __syncthreads();
  for (int o = 128; o > 0; o >>= 1){ if (tid < o) s_red[tid] += s_red[tid+o]; __syncthreads(); }
  if (tid == 0){ float x = s_red[0] + b2[0]; out[b] = 1.f/(1.f + __expf(-x)); }
}

// ---------------- 3-way mixer softmax ----------------
__global__ __launch_bounds__(256) void k_mix(const float* __restrict__ h,
                                             const float* __restrict__ w2,
                                             const float* __restrict__ b2,
                                             float* __restrict__ cwv){
  int b = blockIdx.x, tid = threadIdx.x;
  __shared__ float s_red[256];
  __shared__ float s_sc[3];
  const float* row = h + (size_t)b*DIMD;
  for (int c = 0; c < 3; ++c){
    float p = 0.f;
    for (int j = tid; j < DIMD; j += 256) p = fmaf(row[j], w2[(size_t)c*DIMD + j], p);
    s_red[tid] = p; __syncthreads();
    for (int o = 128; o > 0; o >>= 1){ if (tid < o) s_red[tid] += s_red[tid+o]; __syncthreads(); }
    if (tid == 0) s_sc[c] = s_red[0] + b2[c];
    __syncthreads();
  }
  if (tid == 0){
    float m = fmaxf(s_sc[0], fmaxf(s_sc[1], s_sc[2]));
    float e0 = __expf(s_sc[0]-m), e1 = __expf(s_sc[1]-m), e2 = __expf(s_sc[2]-m);
    float inv = 1.f/(e0+e1+e2);
    cwv[(size_t)b*3+0] = e0*inv; cwv[(size_t)b*3+1] = e1*inv; cwv[(size_t)b*3+2] = e2*inv;
  }
}

// ---------------- state update: W, history, Wt_sequential, ctx_read, write input ----------------
__global__ __launch_bounds__(256) void k_update(
  const float* __restrict__ gkb_, const float* __restrict__ gmem_,
  const float* __restrict__ rvi, float* __restrict__ wts,
  const float* __restrict__ readv, const float* __restrict__ readh,
  const float* __restrict__ cwv, const float* __restrict__ mem,
  float* __restrict__ hist, bf16* __restrict__ wr_in)
{
  int b = blockIdx.x, tid = threadIdx.x;
  __shared__ float s_W[KSLOT];
  float gkb = gkb_[b], gmem = gmem_[b];
  if (tid == 0){
    float w_old[KSLOT];
    for (int k = 0; k < KSLOT; ++k) w_old[k] = wts[(size_t)b*KSLOT + k];
    float first = (1.f - gmem)*gkb;
    for (int k = 0; k < KSLOT; ++k){
      s_W[k] = (gmem*rvi[(size_t)b*KSLOT + k] + w_old[k]*(1.f - gmem))*gkb;
      // wts @ roll(eye,-1,axis=1): new[j] = old[(j+1)%K]*first + old[j]*(1-first)
      wts[(size_t)b*KSLOT + k] = w_old[(k+1)&(KSLOT-1)]*first + w_old[k]*(1.f - first);
    }
  }
  __syncthreads();
  float c0 = cwv[(size_t)b*3+0], c1 = cwv[(size_t)b*3+1], c2 = cwv[(size_t)b*3+2];
  for (int dd = tid; dd < DIMD; dd += 256){
    float rd = readv[(size_t)b*DIMD + dd], rh = readh[(size_t)b*DIMD + dd];
    float now = gkb*rd, last = gmem*rh;
    float latest = (1.f - gkb)*last + now;
    float cr = c0*now + c1*last + c2*latest;
    wr_in[(size_t)b*(2*DIMD) + dd]        = to_bf16(cr);
    wr_in[(size_t)b*(2*DIMD) + DIMD + dd] = to_bf16(mem[(size_t)b*DIMD + dd]);
    float* hrow = hist + ((size_t)b*DIMD + dd)*KSLOT;
    #pragma unroll
    for (int k = 0; k < KSLOT; ++k){
      float h = hrow[k];
      hrow[k] = h*(1.f - s_W[k]) + rd*s_W[k];
    }
  }
}

// =====================================================================
extern "C" void kernel_launch(void* const* d_in, const int* in_sizes, int n_in,
                              void* d_out, int out_size, void* d_ws, size_t ws_size,
                              hipStream_t stream)
{
  (void)in_sizes; (void)n_in; (void)out_size; (void)ws_size;
  const float* in_context = (const float*)d_in[0];
  const float* in_question= (const float*)d_in[1];
  const float* in_knowl   = (const float*)d_in[2];
  const float* in_kbproj  = (const float*)d_in[3];
  const float* in_control = (const float*)d_in[4];
  const float* in_memory  = (const float*)d_in[5];
  const float* in_history = (const float*)d_in[6];
  const float* in_wtseq   = (const float*)d_in[7];
  const float* w_pos   = (const float*)d_in[8];
  const float* b_pos   = (const float*)d_in[9];
  const float* w_cq    = (const float*)d_in[10];
  const float* b_cq    = (const float*)d_in[11];
  const float* w_cattn = (const float*)d_in[12];
  const float* w_rdmem = (const float*)d_in[14];
  const float* b_rdmem = (const float*)d_in[15];
  const float* w_rdcat = (const float*)d_in[16];
  const float* w_rattn = (const float*)d_in[18];
  const float* w_rmmem = (const float*)d_in[20];
  const float* b_rmmem = (const float*)d_in[21];
  const float* w_rmcat = (const float*)d_in[22];
  const float* w_rmattn= (const float*)d_in[24];
  const float* w_write = (const float*)d_in[26];
  const float* b_write = (const float*)d_in[27];
  const float* w_lr1   = (const float*)d_in[28];
  const float* b_lr1   = (const float*)d_in[29];
  const float* w_lr2   = (const float*)d_in[30];
  const float* b_lr2   = (const float*)d_in[31];
  const float* w_lrh1  = (const float*)d_in[32];
  const float* b_lrh1  = (const float*)d_in[33];
  const float* w_lrh2  = (const float*)d_in[34];
  const float* b_lrh2  = (const float*)d_in[35];
  const float* w_mix1  = (const float*)d_in[36];
  const float* b_mix1  = (const float*)d_in[37];
  const float* w_mix2  = (const float*)d_in[38];
  const float* b_mix2  = (const float*)d_in[39];
  // (ctrl_attn_b / read_cat_b / read_attn_b / rm_cat_b / rm_attn_b cancel in softmax)

  char* base = (char*)d_ws; size_t off = 0;
  auto alloc = [&](size_t bytes)->void*{
    off = (off + 255) & ~(size_t)255; void* r = base + off; off += bytes; return r;
  };
  const int D = DIMD, D2 = 2*DIMD, B = BATCH;
  bf16* wb_pos   = (bf16*)alloc((size_t)NSTEP*D*D2*2);
  bf16* wb_cq    = (bf16*)alloc((size_t)D*D2*2);
  bf16* wb_rdm   = (bf16*)alloc((size_t)D*D*2);
  bf16* wb_rmm   = (bf16*)alloc((size_t)D*D*2);
  bf16* wb_catT  = (bf16*)alloc((size_t)D2*D*2);
  bf16* wb_rmcatT= (bf16*)alloc((size_t)D2*D*2);
  bf16* wb_lr1   = (bf16*)alloc((size_t)D2*D2*2);
  bf16* wb_lrh1  = (bf16*)alloc((size_t)D2*D2*2);
  bf16* wb_mix1  = (bf16*)alloc((size_t)D*D*2);
  bf16* wb_wr    = (bf16*)alloc((size_t)D*D2*2);
  bf16* qb       = (bf16*)alloc((size_t)B*D2*2);
  bf16* cq_in    = (bf16*)alloc((size_t)B*D2*2);
  bf16* lr_in    = (bf16*)alloc((size_t)B*D2*2);
  bf16* lrh_in   = (bf16*)alloc((size_t)B*D2*2);
  bf16* wr_in    = (bf16*)alloc((size_t)B*D2*2);
  bf16* mem_bf   = (bf16*)alloc((size_t)B*D*2);
  bf16* vbuf     = (bf16*)alloc((size_t)B*D*2);
  float* cqi     = (float*)alloc((size_t)B*D*4);
  float* control = (float*)alloc((size_t)B*D*4);
  float* mem     = (float*)alloc((size_t)B*D*4);
  float* mem_p   = (float*)alloc((size_t)B*D*4);
  float* memh    = (float*)alloc((size_t)B*D*4);
  float* ubuf    = (float*)alloc((size_t)B*D2*4);
  float* uhbuf   = (float*)alloc((size_t)B*D2*4);
  float* readv   = (float*)alloc((size_t)B*D*4);
  float* readh   = (float*)alloc((size_t)B*D*4);
  float* h1      = (float*)alloc((size_t)B*D2*4);
  float* h2      = (float*)alloc((size_t)B*D2*4);
  float* mixh    = (float*)alloc((size_t)B*D*4);
  float* gkb     = (float*)alloc((size_t)B*4);
  float* gmem    = (float*)alloc((size_t)B*4);
  float* rvi     = (float*)alloc((size_t)B*KSLOT*4);
  float* wts     = (float*)alloc((size_t)B*KSLOT*4);
  float* cwv     = (float*)alloc((size_t)B*3*4);
  float* hist    = (float*)alloc((size_t)B*D*KSLOT*4);

  auto cvt = [&](bf16* dst, const float* src, int n){
    k_cvt_bf16<<<(n+255)/256, 256, 0, stream>>>(dst, src, n);
  };
  auto gemm = [&](const bf16* A, int lda, const bf16* W, const float* bias,
                  float* Cf, bf16* Cb, int ldc, int M, int N, int K, int act){
    int strips = (M/16)*(N/64);
    k_gemm_bf16<<<(strips+7)/8, 256, 0, stream>>>(A, lda, W, bias, Cf, Cb, ldc, M, N, K, act);
  };

  // ---- init: mutable state copies + bf16 weight conversion ----
  (void)hipMemcpyAsync(mem,  in_memory,  (size_t)B*D*4,       hipMemcpyDeviceToDevice, stream);
  (void)hipMemcpyAsync(hist, in_history, (size_t)B*D*KSLOT*4, hipMemcpyDeviceToDevice, stream);
  (void)hipMemcpyAsync(wts,  in_wtseq,   (size_t)B*KSLOT*4,   hipMemcpyDeviceToDevice, stream);
  cvt(wb_pos,  w_pos,   NSTEP*D*D2);
  cvt(wb_cq,   w_cq,    D*D2);
  cvt(wb_rdm,  w_rdmem, D*D);
  cvt(wb_rmm,  w_rmmem, D*D);
  k_cvt_bf16_T<<<(D*D2+255)/256, 256, 0, stream>>>(wb_catT,   w_rdcat, D, D2);
  k_cvt_bf16_T<<<(D*D2+255)/256, 256, 0, stream>>>(wb_rmcatT, w_rmcat, D, D2);
  cvt(wb_lr1,  w_lr1,   D2*D2);
  cvt(wb_lrh1, w_lrh1,  D2*D2);
  cvt(wb_mix1, w_mix1,  D*D);
  cvt(wb_wr,   w_write, D*D2);
  cvt(qb,      in_question, B*D2);
  cvt(mem_bf,  in_memory,   B*D);
  k_cvt_bf16_strided<<<(B*D+255)/256, 256, 0, stream>>>(cq_in, D2, in_control, D, B, D);

  // ---- 4 reasoning steps ----
  for (int i = 0; i < NSTEP; ++i){
    // pa = question @ ctrl_pos_w[i]^T + b -> bf16 into cq_in[:, D:2D]
    gemm(qb, D2, wb_pos + (size_t)i*D*D2, b_pos + (size_t)i*D,
         nullptr, cq_in + D, D2, B, D, D2, 0);
    // cqi = [control, pa] @ ctrl_cq_w^T + b
    gemm(cq_in, D2, wb_cq, b_cq, cqi, nullptr, D, B, D, D2, 0);
    // control = softmax-attended context; bf16 mirrors into cq_in/lr_in/lrh_in first halves
    k_ctx_attn<<<B, 256, 0, stream>>>(cqi, in_context, w_cattn, control, cq_in, lr_in, lrh_in);
    // memory projections
    gemm(mem_bf, D, wb_rdm, b_rdmem, mem_p, nullptr, D, B, D, D, 0);
    gemm(mem_bf, D, wb_rmm, b_rmmem, memh,  nullptr, D, B, D, D, 0);
    // KB branch: u = (control*read_attn_w) @ read_cat_w  (via transposed bf16 copy)
    k_make_v<<<(B*D+255)/256, 256, 0, stream>>>(vbuf, control, w_rattn, B*D);
    gemm(vbuf, D, wb_catT, nullptr, ubuf, nullptr, D2, B, D2, D, 0);
    k_kb_attn<<<B, 256, 0, stream>>>(ubuf, mem_p, in_kbproj, in_knowl, readv, lr_in + D);
    // history branch
    k_make_v<<<(B*D+255)/256, 256, 0, stream>>>(vbuf, control, w_rmattn, B*D);
    gemm(vbuf, D, wb_rmcatT, nullptr, uhbuf, nullptr, D2, B, D2, D, 0);
    k_hist_attn<<<B, 256, 0, stream>>>(uhbuf, memh, hist, rvi, readh, lrh_in + D);
    // gates
    gemm(lr_in,  D2, wb_lr1,  b_lr1,  h1, nullptr, D2, B, D2, D2, 1);
    k_gate<<<B, 256, 0, stream>>>(h1, w_lr2,  b_lr2,  gkb,  D2);
    gemm(lrh_in, D2, wb_lrh1, b_lrh1, h2, nullptr, D2, B, D2, D2, 1);
    k_gate<<<B, 256, 0, stream>>>(h2, w_lrh2, b_lrh2, gmem, D2);
    // mixer (control bf16 lives in lr_in[:, :D] with lda = 2D)
    gemm(lr_in, D2, wb_mix1, b_mix1, mixh, nullptr, D, B, D, D, 1);
    k_mix<<<B, 256, 0, stream>>>(mixh, w_mix2, b_mix2, cwv);
    // state update + write-input assembly
    k_update<<<B, 256, 0, stream>>>(gkb, gmem, rvi, wts, readv, readh, cwv, mem, hist, wr_in);
    // memory = [ctx_read, memory] @ write_w^T + b  (f32 + bf16 mirror)
    gemm(wr_in, D2, wb_wr, b_write, mem, mem_bf, D, B, D, D2, 0);
  }

  (void)hipMemcpyAsync(d_out, mem, (size_t)B*D*4, hipMemcpyDeviceToDevice, stream);
}
